// TemporalAttentionLayer_56487409877348
// MI455X (gfx1250) — compile-verified, matrix operand order checked by reading, not run on hardware
//
#include <hip/hip_runtime.h>
#include <hip/hip_bf16.h>

typedef __attribute__((ext_vector_type(16))) _Float16 v16h;
typedef __attribute__((ext_vector_type(8)))  _Float16 v8h;
typedef __attribute__((ext_vector_type(8)))  float    v8f;

#define BATCH 32768
#define NNB   20
#define NF    172
#define TF    100
#define EDIM  272   // query dim
#define KDIM  444   // key dim
#define HD    136   // head dim

// ---------------------------------------------------------------------------
// Generic f16 WMMA GEMM: C[M,N] = A[M,K] * B[K,N], B given transposed as
// BT[N_padded][K_padded] (row-major, zero-padded in K beyond Kvalid).
// One wave per 16x16 output tile; 4 waves per block along N.
// Epilogue: +bias, optional per-row zeroing, optional ReLU, f32 and/or f16 out.
// ---------------------------------------------------------------------------
__global__ void gemm_f16_wmma(const _Float16* __restrict__ A, int lda,
                              const _Float16* __restrict__ BT, int ldb,
                              int K, int Nvalid,
                              const float* __restrict__ bias,
                              const int* __restrict__ zero_rows,
                              int relu,
                              float* __restrict__ Cf, int ldc,
                              _Float16* __restrict__ Ch, int ldch)
{
    const int lane = threadIdx.x & 31;
    const int wave = threadIdx.x >> 5;
    const int tilesN = (Nvalid + 15) >> 4;
    const int tn = blockIdx.x * 4 + wave;
    if (tn >= tilesN) return;
    const int tm = blockIdx.y;

    const int half = lane >> 4;     // selects K-subgroup per ISA 16-bit layout
    const int qq   = lane & 15;     // M index (A) / N index (B,D)

    const _Float16* arow = A  + (size_t)(tm * 16 + qq) * lda;
    const _Float16* brow = BT + (size_t)(tn * 16 + qq) * ldb;

    v8f acc = {};
    for (int k = 0; k < K; k += 32) {
        const int kb = k + half * 8;
        // ISA 7.12.2: lane half h holds K = h*8+{0..7} and h*8+16+{0..7}
        v8h alo = *(const v8h*)(arow + kb);
        v8h ahi = *(const v8h*)(arow + kb + 16);
        v8h blo = *(const v8h*)(brow + kb);
        v8h bhi = *(const v8h*)(brow + kb + 16);
        v16h av = __builtin_shufflevector(alo, ahi, 0,1,2,3,4,5,6,7,8,9,10,11,12,13,14,15);
        v16h bv = __builtin_shufflevector(blo, bhi, 0,1,2,3,4,5,6,7,8,9,10,11,12,13,14,15);
        acc = __builtin_amdgcn_wmma_f32_16x16x32_f16(
            /*neg_a=*/false, av, /*neg_b=*/false, bv,
            /*c_mod=*/(short)0, acc, /*reuse_a=*/false, /*reuse_b=*/false);
    }

    const int col = tn * 16 + qq;
    if (col >= Nvalid) return;
    const float bsv = bias ? bias[col] : 0.0f;
    const int rowbase = tm * 16 + half * 8;
#pragma unroll
    for (int v = 0; v < 8; ++v) {
        const int row = rowbase + v;
        float val = acc[v] + bsv;
        if (zero_rows && zero_rows[row] != 0) val = 0.0f;
        if (relu) val = fmaxf(val, 0.0f);
        if (Cf) Cf[(size_t)row * ldc  + col] = val;
        if (Ch) Ch[(size_t)row * ldch + col] = (_Float16)val;
    }
}

// ---------------------------------------------------------------------------
// Fused per-node attention: one wave per node. Stages kv (20x444, pad 448,
// LDS pitch 452) once, computes scores against qk, masked softmax, writes
// attn_weights, and the attention-weighted kv sum wv (f16) for the V GEMM.
// ---------------------------------------------------------------------------
#define KV_PITCH 452
__global__ void attn_softmax_kernel(const float* __restrict__ nf,
                                    const float* __restrict__ ef,
                                    const float* __restrict__ tfeat,
                                    const int*   __restrict__ mask,
                                    const _Float16* __restrict__ qk, // [B,896]
                                    float* __restrict__ attn_w,      // [B,20]
                                    _Float16* __restrict__ wv,       // [B,896]
                                    int* __restrict__ invalid_flags)
{
    __shared__ float kv[NNB * KV_PITCH];
    __shared__ float qks[896];
    __shared__ float sc[2 * NNB];
    __shared__ float at[2 * NNB];
    __shared__ float red[4];

    const int b = blockIdx.x;
    const int lane = threadIdx.x;

    for (int idx = lane; idx < NNB * 448; idx += 32) {
        const int n = idx / 448;
        const int j = idx - n * 448;
        float v;
        if      (j < NF)        v = nf[((size_t)b * NNB + n) * NF + j];
        else if (j < 2 * NF)    v = ef[((size_t)b * NNB + n) * NF + (j - NF)];
        else if (j < KDIM)      v = tfeat[((size_t)b * NNB + n) * TF + (j - 2 * NF)];
        else                    v = 0.0f;
        kv[n * KV_PITCH + j] = v;
    }
    for (int idx = lane; idx < 896; idx += 32)
        qks[idx] = (float)qk[(size_t)b * 896 + idx];

    bool mk = true;
    if (lane < NNB) mk = (mask[(size_t)b * NNB + lane] != 0);
    const bool invalid = (__all(mk ? 1 : 0) != 0);  // all 20 masked
    bool masked = mk;
    if (invalid && lane == 0) masked = false;       // reference unmask trick

    __syncthreads();

    if (lane < NNB) {
        float s0 = 0.f, s1 = 0.f;
        const float* kr = &kv[lane * KV_PITCH];
        for (int j = 0; j < KDIM; ++j) {
            const float kvv = kr[j];
            s0 += qks[j]       * kvv;
            s1 += qks[448 + j] * kvv;
        }
        const float rs = 0.08574929257125442f;  // 1/sqrt(136)
        s0 = (s0 + qks[KDIM])       * rs;       // col 444 = q . k_b (head 0)
        s1 = (s1 + qks[448 + KDIM]) * rs;
        if (masked) { s0 = -1e30f; s1 = -1e30f; }
        sc[lane] = s0;
        sc[NNB + lane] = s1;
    }
    __syncthreads();
    if (lane < 2) {
        float mx = -3.0e38f;
        for (int n = 0; n < NNB; ++n) mx = fmaxf(mx, sc[lane * NNB + n]);
        float sum = 0.f;
        for (int n = 0; n < NNB; ++n) sum += __expf(sc[lane * NNB + n] - mx);
        red[lane] = mx;
        red[2 + lane] = 1.0f / sum;
    }
    __syncthreads();
    if (lane < NNB) {
        const float a0 = __expf(sc[lane]       - red[0]) * red[2];
        const float a1 = __expf(sc[NNB + lane] - red[1]) * red[3];
        at[lane] = a0;
        at[NNB + lane] = a1;
        attn_w[(size_t)b * NNB + lane] = 0.5f * (a0 + a1);
    }
    if (lane == 0) invalid_flags[b] = invalid ? 1 : 0;
    __syncthreads();

    for (int j = lane; j < 448; j += 32) {
        float w0 = 0.f, w1 = 0.f;
        if (j < KDIM) {
            for (int n = 0; n < NNB; ++n) {
                const float kvv = kv[n * KV_PITCH + j];
                w0 += at[n]       * kvv;
                w1 += at[NNB + n] * kvv;
            }
        }
        wv[(size_t)b * 896 + j]       = (_Float16)w0;
        wv[(size_t)b * 896 + 448 + j] = (_Float16)w1;
    }
}

// ---------------------------------------------------------------------------
// Packing kernels (f32 weights -> padded f16 BT layouts, input staging)
// ---------------------------------------------------------------------------
__global__ void pack_bt_kernel(const float* __restrict__ W, _Float16* __restrict__ BT,
                               int Nv, int Kv, int rows, int ldb)
{
    const int idx = blockIdx.x * 256 + threadIdx.x;
    if (idx >= rows * ldb) return;
    const int n = idx / ldb;
    const int k = idx - n * ldb;
    const float v = (n < Nv && k < Kv) ? W[(size_t)n * Kv + k] : 0.0f;
    BT[idx] = (_Float16)v;
}

// BT_k[h][n][d] = d<136 ? (n<444 ? k_w[h*136+d][n] : n==444 ? k_b[h*136+d] : 0) : 0
__global__ void pack_btk_kernel(const float* __restrict__ k_w,
                                const float* __restrict__ k_b,
                                _Float16* __restrict__ BT)
{
    const int idx = blockIdx.x * 256 + threadIdx.x;
    if (idx >= 2 * 448 * 160) return;
    const int h = idx / (448 * 160);
    const int r = idx - h * (448 * 160);
    const int n = r / 160;
    const int d = r - n * 160;
    float v = 0.0f;
    if (d < HD) {
        if (n < KDIM)       v = k_w[(size_t)(h * HD + d) * KDIM + n];
        else if (n == KDIM) v = k_b[h * HD + d];
    }
    BT[idx] = (_Float16)v;
}

__global__ void pack_query_kernel(const float* __restrict__ sn,
                                  const float* __restrict__ st,
                                  _Float16* __restrict__ q)
{
    const int idx = blockIdx.x * 256 + threadIdx.x;
    if (idx >= BATCH * 288) return;
    const int b = idx / 288;
    const int j = idx - b * 288;
    float v = 0.0f;
    if      (j < NF)   v = sn[(size_t)b * NF + j];
    else if (j < EDIM) v = st[(size_t)b * TF + (j - NF)];
    q[idx] = (_Float16)v;
}

// merged[b][272 + i] = src_node[b][i]; merged[b][444..447] = 0
__global__ void pack_merged_kernel(const float* __restrict__ sn,
                                   _Float16* __restrict__ merged)
{
    const int idx = blockIdx.x * 256 + threadIdx.x;
    if (idx >= BATCH * 176) return;
    const int b = idx / 176;
    const int i = idx - b * 176;
    const float v = (i < NF) ? sn[(size_t)b * NF + i] : 0.0f;
    merged[(size_t)b * 448 + EDIM + i] = (_Float16)v;
}

extern "C" void kernel_launch(void* const* d_in, const int* in_sizes, int n_in,
                              void* d_out, int out_size, void* d_ws, size_t ws_size,
                              hipStream_t stream) {
    const float* src_node = (const float*)d_in[0];
    const float* src_time = (const float*)d_in[1];
    const float* nbr_feat = (const float*)d_in[2];
    const float* nbr_time = (const float*)d_in[3];
    const float* edge_ft  = (const float*)d_in[4];
    const int*   mask     = (const int*)d_in[5];
    const float* q_w  = (const float*)d_in[6];
    const float* k_w  = (const float*)d_in[7];
    const float* v_w  = (const float*)d_in[8];
    const float* q_b  = (const float*)d_in[9];
    const float* k_b  = (const float*)d_in[10];
    const float* v_b  = (const float*)d_in[11];
    const float* o_w  = (const float*)d_in[12];
    const float* o_b  = (const float*)d_in[13];
    const float* m1_w = (const float*)d_in[14];
    const float* m1_b = (const float*)d_in[15];
    const float* m2_w = (const float*)d_in[16];
    const float* m2_b = (const float*)d_in[17];
    float* out = (float*)d_out;

    const size_t B = BATCH;
    char* p = (char*)d_ws;
    auto take = [&](size_t bytes) { char* r = p; p += (bytes + 255) & ~(size_t)255; return r; };

    _Float16* query_h  = (_Float16*)take(B * 288 * 2);
    _Float16* q_h      = (_Float16*)take(B * 320 * 2);   // 272 valid, K-pad slack
    _Float16* qk_h     = (_Float16*)take(B * 896 * 2);   // [B][2][448], col444=bias term
    _Float16* wv_h     = (_Float16*)take(B * 896 * 2);
    _Float16* ctx_h    = (_Float16*)take(B * 288 * 2);
    _Float16* merged_h = (_Float16*)take(B * 448 * 2);
    _Float16* hidden_h = (_Float16*)take(B * 192 * 2);
    int*      invalid  = (int*)take(B * 4);
    _Float16* bt_q   = (_Float16*)take((size_t)272 * 288 * 2);
    _Float16* bt_k   = (_Float16*)take((size_t)2 * 448 * 160 * 2);
    _Float16* bt_v   = (_Float16*)take((size_t)2 * 144 * 448 * 2);
    _Float16* bt_out = (_Float16*)take((size_t)272 * 288 * 2);
    _Float16* bt_m1  = (_Float16*)take((size_t)176 * 448 * 2);
    _Float16* bt_m2  = (_Float16*)take((size_t)176 * 192 * 2);

    const int MT = BATCH / 16;  // 2048 M tiles
    auto gb = [](int elems) { return (elems + 255) / 256; };

    // --- pack weights / inputs ---
    pack_bt_kernel<<<gb(272 * 288), 256, 0, stream>>>(q_w,  bt_q,  272, 272, 272, 288);
    pack_bt_kernel<<<gb(272 * 288), 256, 0, stream>>>(o_w,  bt_out,272, 272, 272, 288);
    pack_bt_kernel<<<gb(176 * 448), 256, 0, stream>>>(m1_w, bt_m1, 172, 444, 176, 448);
    pack_bt_kernel<<<gb(176 * 192), 256, 0, stream>>>(m2_w, bt_m2, 172, 172, 176, 192);
    pack_bt_kernel<<<gb(144 * 448), 256, 0, stream>>>(v_w,                 bt_v,             136, 444, 144, 448);
    pack_bt_kernel<<<gb(144 * 448), 256, 0, stream>>>(v_w + (size_t)HD * KDIM, bt_v + (size_t)144 * 448, 136, 444, 144, 448);
    pack_btk_kernel<<<gb(2 * 448 * 160), 256, 0, stream>>>(k_w, k_b, bt_k);
    pack_query_kernel<<<gb(BATCH * 288), 256, 0, stream>>>(src_node, src_time, query_h);
    pack_merged_kernel<<<gb(BATCH * 176), 256, 0, stream>>>(src_node, merged_h);

    // --- q = query @ q_w.T + q_b  (f16 out for next stage) ---
    gemm_f16_wmma<<<dim3(5, MT), 128, 0, stream>>>(
        query_h, 288, bt_q, 288, 288, 272, q_b, nullptr, 0,
        nullptr, 0, q_h, 320);

    // --- qk[h] = q_h @ k_w_h  (N=445: col 444 carries q.k_b) ---
    for (int h = 0; h < 2; ++h) {
        gemm_f16_wmma<<<dim3(7, MT), 128, 0, stream>>>(
            q_h + h * HD, 320, bt_k + (size_t)h * 448 * 160, 160, 160, 445,
            nullptr, nullptr, 0, nullptr, 0, qk_h + h * 448, 896);
    }

    // --- fused attention: scores, softmax, attn_weights, wv ---
    attn_softmax_kernel<<<BATCH, 32, 0, stream>>>(
        nbr_feat, edge_ft, nbr_time, mask, qk_h,
        out + (size_t)BATCH * NF, wv_h, invalid);

    // --- ctx[h] = wv[h] @ v_w_h.T + v_b[h]  (softmax sums to 1 => v_b is plain bias) ---
    for (int h = 0; h < 2; ++h) {
        gemm_f16_wmma<<<dim3(3, MT), 128, 0, stream>>>(
            wv_h + h * 448, 896, bt_v + (size_t)h * 144 * 448, 448, 448, 136,
            v_b + h * HD, nullptr, 0, nullptr, 0, ctx_h + h * HD, 288);
    }

    // --- attn_out = ctx @ out_w.T + out_b, zeroed for invalid rows, into merged[:,0:272] ---
    gemm_f16_wmma<<<dim3(5, MT), 128, 0, stream>>>(
        ctx_h, 288, bt_out, 288, 288, 272, o_b, invalid, 0,
        nullptr, 0, merged_h, 448);

    // --- hidden = relu(merged @ m1_w.T + m1_b) ---
    gemm_f16_wmma<<<dim3(3, MT), 128, 0, stream>>>(
        merged_h, 448, bt_m1, 448, 448, 172, m1_b, nullptr, 1,
        nullptr, 0, hidden_h, 192);

    // --- output = hidden @ m2_w.T + m2_b  (f32 to d_out) ---
    gemm_f16_wmma<<<dim3(3, MT), 128, 0, stream>>>(
        hidden_h, 192, bt_m2, 192, 192, 172, m2_b, nullptr, 0,
        out, 172, nullptr, 0);

    (void)in_sizes; (void)n_in; (void)out_size; (void)ws_size;
}
